// Attention_48661979463745
// MI455X (gfx1250) — compile-verified, matrix operand order checked by reading, not run on hardware
//
#include <hip/hip_runtime.h>
#include <hip/hip_bf16.h>
#include <stdint.h>

// MI455X / gfx1250: wave32, WMMA 16x16x32 bf16 + async global->LDS staging.
typedef __attribute__((ext_vector_type(16))) __bf16 v16bf;
typedef __attribute__((ext_vector_type(8)))  float  v8f;

#define B_   2
#define S_   2048
#define D_   4096
#define NH_  32
#define NKV_ 8
#define HD_  128
#define NREP_ 4

static __device__ __forceinline__ unsigned short f2bf(float f) {
    unsigned u = __float_as_uint(f);
    unsigned r = u + 0x7FFFu + ((u >> 16) & 1u);   // round-to-nearest-even
    return (unsigned short)(r >> 16);
}
static __device__ __forceinline__ float bf2f(unsigned short h) {
    return __uint_as_float(((unsigned)h) << 16);
}

// LDS byte offset of a __shared__ object: generic pointers into the LDS
// aperture carry the AS3 offset in their low 32 bits.
static __device__ __forceinline__ unsigned lds_off(const void* p) {
    return (unsigned)(size_t)p;
}

// Async copy: 16 bytes global -> LDS per lane, tracked by ASYNCcnt.
static __device__ __forceinline__ void async_b128(unsigned ldsoff, const void* g) {
    unsigned long long ga = (unsigned long long)g;
    asm volatile("global_load_async_to_lds_b128 %0, %1, off"
                 :: "v"(ldsoff), "v"(ga) : "memory");
}
static __device__ __forceinline__ void wait_async0() {
    asm volatile("s_wait_asynccnt 0x0" ::: "memory");
}

union FragAB { v16bf v; unsigned u[8]; };
union FragC  { v8f   v; float    f[8]; };

// ---------------------------------------------------------------------------
// fp32 -> bf16 conversion (grid-stride)
// ---------------------------------------------------------------------------
__global__ void f32_to_bf16_kernel(const float* __restrict__ src,
                                   unsigned short* __restrict__ dst, int n) {
    for (int i = blockIdx.x * blockDim.x + threadIdx.x; i < n;
         i += gridDim.x * blockDim.x)
        dst[i] = f2bf(src[i]);
}

// ---------------------------------------------------------------------------
// fp32 [K][N] -> bf16 [N][K] transposed conversion (LDS-tiled, 32x32 tiles)
// ---------------------------------------------------------------------------
__global__ __launch_bounds__(256)
void f32_to_bf16_transpose_kernel(const float* __restrict__ src,
                                  unsigned short* __restrict__ dst,
                                  int K, int N) {
    __shared__ float tile[32][33];
    int n0 = blockIdx.x * 32, k0 = blockIdx.y * 32;
    int tx = threadIdx.x & 31, ty = threadIdx.x >> 5;   // 32 x 8
#pragma unroll
    for (int i = 0; i < 32; i += 8)
        tile[ty + i][tx] = src[(size_t)(k0 + ty + i) * N + n0 + tx];
    __syncthreads();
#pragma unroll
    for (int i = 0; i < 32; i += 8)
        dst[(size_t)(n0 + ty + i) * K + k0 + tx] = f2bf(tile[tx][ty + i]);
}

// ---------------------------------------------------------------------------
// bf16 GEMM: C[M,N] = A[M,K] * Wt[N,K]^T  (both row-major, Wt pre-transposed).
// Block tile 128x128, K-step 32. 8 waves: 4 (M) x 2 (N), each wave 32x64.
// Tiles staged with GLOBAL_LOAD_ASYNC_TO_LDS_B128 (ASYNCcnt). LDS stored as
// packed bf16 K-pairs; WMMA fragment loads are single 32-bit ds reads:
//   frag.u[v] = tile[row][ (v>=4)*8 + half*4 + (v&3) ]   (K = 2*idx, 2*idx+1)
// Row pad = 20 uints (80B): 16B-aligned chunks, conflict-free columns.
// ---------------------------------------------------------------------------
__global__ __launch_bounds__(256)
void gemm_bf16_wmma(const unsigned short* __restrict__ A,
                    const unsigned short* __restrict__ Wt,
                    unsigned short* __restrict__ Cb,
                    float* __restrict__ Cf,
                    int M, int N, int K, int storeF32) {
    __shared__ unsigned As[128][20];   // [m][k/2]
    __shared__ unsigned Bs[128][20];   // [n][k/2]

    const int tid  = threadIdx.x;
    const int wid  = tid >> 5;
    const int lane = tid & 31;
    const int half = lane >> 4;
    const int lm   = lane & 15;

    const int m0    = blockIdx.y * 128;
    const int n0    = blockIdx.x * 128;
    const int waveM = (wid >> 1) * 32;
    const int waveN = (wid & 1) * 64;

    const unsigned asBase = lds_off(&As[0][0]);
    const unsigned bsBase = lds_off(&Bs[0][0]);

    FragC acc[2][4];
#pragma unroll
    for (int i = 0; i < 2; ++i)
#pragma unroll
        for (int j = 0; j < 4; ++j)
#pragma unroll
            for (int e = 0; e < 8; ++e) acc[i][j].f[e] = 0.0f;

    for (int k0 = 0; k0 < K; k0 += 32) {
        // stage A and B tiles: 128 rows x 64B each = 512 16B-chunks per tile
#pragma unroll
        for (int p = 0; p < 2; ++p) {
            int rr = (tid >> 2) + p * 64;
            int cc = tid & 3;
            async_b128(asBase + rr * 80 + cc * 16,
                       A  + (size_t)(m0 + rr) * K + k0 + cc * 8);
            async_b128(bsBase + rr * 80 + cc * 16,
                       Wt + (size_t)(n0 + rr) * K + k0 + cc * 8);
        }
        wait_async0();
        __syncthreads();

        FragAB af[2], bf4[4];
#pragma unroll
        for (int mt = 0; mt < 2; ++mt) {
            int row = waveM + mt * 16 + lm;
#pragma unroll
            for (int v = 0; v < 8; ++v)
                af[mt].u[v] = As[row][((v >> 2) << 3) + (half << 2) + (v & 3)];
        }
#pragma unroll
        for (int nt = 0; nt < 4; ++nt) {
            int col = waveN + nt * 16 + lm;
#pragma unroll
            for (int v = 0; v < 8; ++v)
                bf4[nt].u[v] = Bs[col][((v >> 2) << 3) + (half << 2) + (v & 3)];
        }
#pragma unroll
        for (int mt = 0; mt < 2; ++mt)
#pragma unroll
            for (int nt = 0; nt < 4; ++nt)
                acc[mt][nt].v = __builtin_amdgcn_wmma_f32_16x16x32_bf16(
                    false, af[mt].v, false, bf4[nt].v,
                    (short)0, acc[mt][nt].v, false, false);
        __syncthreads();
    }

    // epilogue: C layout row M = v + 8*half, col N = lane%16
#pragma unroll
    for (int mt = 0; mt < 2; ++mt)
#pragma unroll
        for (int nt = 0; nt < 4; ++nt)
#pragma unroll
            for (int v = 0; v < 8; ++v) {
                int row = m0 + waveM + mt * 16 + v + 8 * half;
                int col = n0 + waveN + nt * 16 + lm;
                if (storeF32) Cf[(size_t)row * N + col] = acc[mt][nt].f[v];
                else          Cb[(size_t)row * N + col] = f2bf(acc[mt][nt].f[v]);
            }
}

// ---------------------------------------------------------------------------
// RoPE on bf16 [B*S, nheads*HD] in-place (interleaved pairs)
// ---------------------------------------------------------------------------
__global__ void rope_kernel(unsigned short* __restrict__ qk,
                            const float* __restrict__ fc,
                            const float* __restrict__ fs,
                            int nheads, int total_pairs) {
    int i = blockIdx.x * blockDim.x + threadIdx.x;
    if (i >= total_pairs) return;
    int d = i % (HD_ / 2);
    int t = i / (HD_ / 2);
    int h = t % nheads;
    int row = t / nheads;                  // b*S + s
    int s = row % S_;
    float c  = fc[s * (HD_ / 2) + d];
    float sn = fs[s * (HD_ / 2) + d];
    size_t base = (size_t)row * nheads * HD_ + (size_t)h * HD_ + 2 * d;
    float xr = bf2f(qk[base]), xi = bf2f(qk[base + 1]);
    qk[base]     = f2bf(xr * c - xi * sn);
    qk[base + 1] = f2bf(xr * sn + xi * c);
}

// ---------------------------------------------------------------------------
// Flash attention (causal, GQA). One WG = one (b,h) x 128-query block.
// 8 waves, wave w owns 16 query rows. Key tiles of 32.
// Q/K tiles staged with async global->LDS; V transposed manually.
// Per tile/wave: 8 WMMA (QK^T over HD=128) + 8 WMMA (PV into 16x128 acc).
// ---------------------------------------------------------------------------
__global__ __launch_bounds__(256)
void flash_attn_kernel(const unsigned short* __restrict__ Q,
                       const unsigned short* __restrict__ Kb,
                       const unsigned short* __restrict__ Vb,
                       unsigned short* __restrict__ O) {
    __shared__ unsigned Qs[128][68];      // [qrow][hd/2], 272B rows (16B align)
    __shared__ unsigned Ks[32][68];       // [key][hd/2]
    __shared__ unsigned Vs[128][17];      // [hd][key/2] (transposed)
    __shared__ unsigned Ps[8][16][16];    // per-wave P tile, [row][key/2]

    const int tid = threadIdx.x, wid = tid >> 5, lane = tid & 31;
    const int half = lane >> 4, lm = lane & 15;
    const int q0 = blockIdx.x * 128;
    const int bh = blockIdx.y;
    const int b = bh / NH_, h = bh % NH_, hk = h / NREP_;

    const unsigned qsBase = lds_off(&Qs[0][0]);
    const unsigned ksBase = lds_off(&Ks[0][0]);

    // stage Q block async: 128 rows x 256B = 2048 16B-chunks
#pragma unroll
    for (int p = 0; p < 8; ++p) {
        int rr = (tid >> 4) + p * 16;
        int cc = tid & 15;
        async_b128(qsBase + rr * 272 + cc * 16,
                   Q + (size_t)(b * S_ + q0 + rr) * (NH_ * HD_) + h * HD_ + cc * 8);
    }

    FragC acc[8];
    float m_s[8], l_s[8];
#pragma unroll
    for (int nt = 0; nt < 8; ++nt)
#pragma unroll
        for (int e = 0; e < 8; ++e) acc[nt].f[e] = 0.0f;
#pragma unroll
    for (int e = 0; e < 8; ++e) { m_s[e] = -1e30f; l_s[e] = 0.0f; }

    const float scale = 0.08838834764831845f;   // 1/sqrt(128)
    const int qr = wid * 16;

    for (int kt = 0; kt < q0 + 128; kt += 32) {
        __syncthreads();                        // LDS reuse fence
        // stage K tile async: 32 rows x 256B = 512 chunks
#pragma unroll
        for (int p = 0; p < 2; ++p) {
            int rr = (tid >> 4) + p * 16;
            int cc = tid & 15;
            async_b128(ksBase + rr * 272 + cc * 16,
                       Kb + (size_t)(b * S_ + kt + rr) * (NKV_ * HD_) + hk * HD_ + cc * 8);
        }
        // stage V tile transposed: Vs[hd][key/2]
#pragma unroll
        for (int it = 0; it < 8; ++it) {
            int idx = tid + it * 256;
            int hd = idx & 127, c = idx >> 7;    // c in 0..15
            unsigned short v0 = Vb[(size_t)(b * S_ + kt + 2 * c    ) * (NKV_ * HD_) + hk * HD_ + hd];
            unsigned short v1 = Vb[(size_t)(b * S_ + kt + 2 * c + 1) * (NKV_ * HD_) + hk * HD_ + hd];
            Vs[hd][c] = (unsigned)v0 | ((unsigned)v1 << 16);
        }
        wait_async0();
        __syncthreads();

        // S = Q*K^T : 16 q-rows x 32 keys (2 N-tiles), K-dim = HD in 4 steps
        FragC sc[2];
#pragma unroll
        for (int nt = 0; nt < 2; ++nt)
#pragma unroll
            for (int e = 0; e < 8; ++e) sc[nt].f[e] = 0.0f;
#pragma unroll
        for (int kk = 0; kk < 4; ++kk) {
            FragAB aq;
            int row = qr + lm;
#pragma unroll
            for (int v = 0; v < 8; ++v)
                aq.u[v] = Qs[row][kk * 16 + ((v >> 2) << 3) + (half << 2) + (v & 3)];
#pragma unroll
            for (int nt = 0; nt < 2; ++nt) {
                FragAB bk;
                int key = nt * 16 + lm;
#pragma unroll
                for (int v = 0; v < 8; ++v)
                    bk.u[v] = Ks[key][kk * 16 + ((v >> 2) << 3) + (half << 2) + (v & 3)];
                sc[nt].v = __builtin_amdgcn_wmma_f32_16x16x32_bf16(
                    false, aq.v, false, bk.v, (short)0, sc[nt].v, false, false);
            }
        }

        // causal mask + online softmax (row M = v + 8*half, key N = lane%16)
        float p0v[8], p1v[8];
#pragma unroll
        for (int v = 0; v < 8; ++v) {
            int qrow = q0 + qr + v + 8 * half;
            float s0 = sc[0].f[v] * scale;
            float s1 = sc[1].f[v] * scale;
            if (kt + lm      > qrow) s0 = -1e9f;
            if (kt + 16 + lm > qrow) s1 = -1e9f;
            float mx = fmaxf(s0, s1);
#pragma unroll
            for (int off = 1; off < 16; off <<= 1)
                mx = fmaxf(mx, __shfl_xor(mx, off, 32));
            float mn = fmaxf(m_s[v], mx);
            float p0 = __expf(s0 - mn), p1 = __expf(s1 - mn);
            float ls = p0 + p1;
#pragma unroll
            for (int off = 1; off < 16; off <<= 1)
                ls += __shfl_xor(ls, off, 32);
            float alpha = __expf(m_s[v] - mn);
            l_s[v] = l_s[v] * alpha + ls;
            m_s[v] = mn;
            p0v[v] = p0; p1v[v] = p1;
#pragma unroll
            for (int nt = 0; nt < 8; ++nt) acc[nt].f[v] *= alpha;
        }

        // P (C-layout) -> LDS -> A-fragment layout
        unsigned short* Ph = (unsigned short*)&Ps[wid][0][0];   // [16][32] bf16
#pragma unroll
        for (int v = 0; v < 8; ++v) {
            Ph[(v + 8 * half) * 32 + lm]      = f2bf(p0v[v]);
            Ph[(v + 8 * half) * 32 + 16 + lm] = f2bf(p1v[v]);
        }
        FragAB ap;
        const unsigned* Pu = &Ps[wid][0][0];
#pragma unroll
        for (int v = 0; v < 8; ++v)
            ap.u[v] = Pu[lm * 16 + ((v >> 2) << 3) + (half << 2) + (v & 3)];

        // O += P * V  (A = P 16x32, B = V 32x128 -> 8 N-tiles)
#pragma unroll
        for (int nt = 0; nt < 8; ++nt) {
            FragAB bv;
            int col = nt * 16 + lm;
#pragma unroll
            for (int v = 0; v < 8; ++v)
                bv.u[v] = Vs[col][((v >> 2) << 3) + (half << 2) + (v & 3)];
            acc[nt].v = __builtin_amdgcn_wmma_f32_16x16x32_bf16(
                false, ap.v, false, bv.v, (short)0, acc[nt].v, false, false);
        }
    }

    // normalize and store bf16 [B*S, NH*HD]
#pragma unroll
    for (int nt = 0; nt < 8; ++nt)
#pragma unroll
        for (int v = 0; v < 8; ++v) {
            int qrow = q0 + qr + v + 8 * half;
            float inv = 1.0f / l_s[v];
            O[(size_t)(b * S_ + qrow) * (NH_ * HD_) + h * HD_ + nt * 16 + lm] =
                f2bf(acc[nt].f[v] * inv);
        }
}

// ---------------------------------------------------------------------------
// Host-side orchestration
// ---------------------------------------------------------------------------
extern "C" void kernel_launch(void* const* d_in, const int* in_sizes, int n_in,
                              void* d_out, int out_size, void* d_ws, size_t ws_size,
                              hipStream_t stream) {
    (void)in_sizes; (void)n_in; (void)out_size; (void)ws_size;
    const float* x  = (const float*)d_in[0];
    // d_in[1] = start_pos (0), d_in[4] = mask (causal handled analytically)
    const float* fc = (const float*)d_in[2];
    const float* fs = (const float*)d_in[3];
    const float* wq = (const float*)d_in[5];
    const float* wk = (const float*)d_in[6];
    const float* wv = (const float*)d_in[7];
    const float* wo = (const float*)d_in[8];

    char* ws = (char*)d_ws;
    size_t off = 0;
    const size_t XN  = (size_t)B_ * S_ * D_;              // 16777216
    const size_t WQN = (size_t)D_ * NH_ * HD_;            // 16777216
    const size_t WKN = (size_t)D_ * NKV_ * HD_;           // 4194304
    unsigned short* xb  = (unsigned short*)(ws + off); off += XN  * 2;
    unsigned short* wqb = (unsigned short*)(ws + off); off += WQN * 2;   // [N][K]
    unsigned short* wkb = (unsigned short*)(ws + off); off += WKN * 2;   // [N][K]
    unsigned short* wvb = (unsigned short*)(ws + off); off += WKN * 2;   // [N][K]
    unsigned short* wob = (unsigned short*)(ws + off); off += WQN * 2;   // [N][K]
    unsigned short* qb  = (unsigned short*)(ws + off); off += XN  * 2;
    unsigned short* kb  = (unsigned short*)(ws + off); off += (size_t)B_ * S_ * NKV_ * HD_ * 2;
    unsigned short* vb  = (unsigned short*)(ws + off); off += (size_t)B_ * S_ * NKV_ * HD_ * 2;
    unsigned short* ab  = (unsigned short*)(ws + off); off += XN  * 2;

    // 1) fp32 -> bf16 (x plain; weights transposed to [N][K])
    f32_to_bf16_kernel<<<4096, 256, 0, stream>>>(x, xb, (int)XN);
    f32_to_bf16_transpose_kernel<<<dim3(4096 / 32, 4096 / 32), 256, 0, stream>>>(wq, wqb, 4096, 4096);
    f32_to_bf16_transpose_kernel<<<dim3(1024 / 32, 4096 / 32), 256, 0, stream>>>(wk, wkb, 4096, 1024);
    f32_to_bf16_transpose_kernel<<<dim3(1024 / 32, 4096 / 32), 256, 0, stream>>>(wv, wvb, 4096, 1024);
    f32_to_bf16_transpose_kernel<<<dim3(4096 / 32, 4096 / 32), 256, 0, stream>>>(wo, wob, 4096, 4096);

    const int Mrows = B_ * S_;   // 4096
    // 2) QKV projections (WMMA bf16, async-LDS staged)
    gemm_bf16_wmma<<<dim3(32, 32), 256, 0, stream>>>(xb, wqb, qb, nullptr, Mrows, 4096, D_, 0);
    gemm_bf16_wmma<<<dim3(8,  32), 256, 0, stream>>>(xb, wkb, kb, nullptr, Mrows, 1024, D_, 0);
    gemm_bf16_wmma<<<dim3(8,  32), 256, 0, stream>>>(xb, wvb, vb, nullptr, Mrows, 1024, D_, 0);

    // 3) RoPE on Q and K
    rope_kernel<<<(8388608 + 255) / 256, 256, 0, stream>>>(qb, fc, fs, NH_,  8388608);
    rope_kernel<<<(2097152 + 255) / 256, 256, 0, stream>>>(kb, fc, fs, NKV_, 2097152);

    // 4) flash attention: grid = (S/128, B*NH)
    flash_attn_kernel<<<dim3(S_ / 128, B_ * NH_), 256, 0, stream>>>(qb, kb, vb, ab);

    // 5) output projection, fp32 result
    gemm_bf16_wmma<<<dim3(32, 32), 256, 0, stream>>>(ab, wob, nullptr, (float*)d_out, Mrows, 4096, D_, 1);
}
